// FusedFuzzyDeepNet_13091060318829
// MI455X (gfx1250) — matrix operand order repaired and checked by reading, not compiled, not run on hardware
//
#include <hip/hip_runtime.h>
#include <hip/hip_bf16.h>

typedef __attribute__((ext_vector_type(16))) _Float16 v16h;
typedef __attribute__((ext_vector_type(8)))  _Float16 v8h;
typedef __attribute__((ext_vector_type(4)))  _Float16 v4h;
typedef __attribute__((ext_vector_type(8)))  float    v8f;

#define BATCH   65536
#define DDIM    256
#define CN      8
#define BN_EPS  1e-5f

// LDS geometry (halves). Row stride padded for bank-conflict-free A-frag loads.
#define RS     264          // activation row stride (528B -> lane m starts at bank 4m)
#define BPS    72           // B-panel row stride (144B -> 16 distinct start banks)
#define SLAB   (16 * RS)    // per-wave 16-row slab
#define PANEL  (256 * BPS)  // one weight panel (double-buffered)

// Workspace float layout: [0..255] sum, [256..511] sumsq, [512..767] scale,
// [768..1023] shift, [1024..1039] padded b_o.  Half region at byte 4352:
//   wfiT[256*256], w1T[256*256], w2T[256*256], wfuT[256*512], woT[16*256]

// ---------------- stats ----------------
__global__ void k_zero_stats(float* ws) {
    ws[threadIdx.x] = 0.f;            // 512 threads: sum + sumsq
}

__global__ void k_colstats(const float* __restrict__ x, float* ws) {
    const int c  = threadIdx.x;       // column
    const int r0 = blockIdx.x * 256;  // 256 blocks x 256 rows
    float s = 0.f, s2 = 0.f;
    for (int r = 0; r < 256; ++r) {
        float v = x[(size_t)(r0 + r) * DDIM + c];
        s += v; s2 += v * v;
    }
    atomicAdd(&ws[c], s);
    atomicAdd(&ws[DDIM + c], s2);
}

__global__ void k_finalize(const float* __restrict__ gamma,
                           const float* __restrict__ beta,
                           const float* __restrict__ b_o, float* ws) {
    const int c = threadIdx.x;
    const float inv = 1.f / (float)BATCH;
    float mean = ws[c] * inv;
    float var  = ws[DDIM + c] * inv - mean * mean;
    float sc   = gamma[c] * rsqrtf(var + BN_EPS);
    ws[512 + c] = sc;
    ws[768 + c] = beta[c] - mean * sc;
    if (c < 16) ws[1024 + c] = (c < CN) ? b_o[c] : 0.f;
}

// ---------------- weight prep: fp32 [K][N] -> f16 [N][K] ----------------
__global__ void k_transpose(const float* __restrict__ src, _Float16* __restrict__ dst,
                            int K, int N, int total) {
    for (int i = blockIdx.x * blockDim.x + threadIdx.x; i < total;
         i += gridDim.x * blockDim.x) {
        int k = i / N, n = i - k * N;
        dst[(size_t)n * K + k] = (_Float16)src[i];
    }
}

__global__ void k_prep_wo(const float* __restrict__ W_o, _Float16* __restrict__ dst) {
    int i = blockIdx.x * blockDim.x + threadIdx.x;   // 0..4095 -> [16][256]
    int n = i >> 8, k = i & 255;
    dst[i] = (n < CN) ? (_Float16)W_o[k * CN + n] : (_Float16)0.f;
}

// ---------------- WMMA fragment loaders ----------------
// A 16x32 f16: lanes 0-15 row m, K 0-15 contiguous; lanes 16-31: K 16-31
__device__ __forceinline__ v16h ldA(const _Float16* p) {
    v8h lo = *(const v8h*)(p);
    v8h hi = *(const v8h*)(p + 8);
    v16h r;
#pragma unroll
    for (int i = 0; i < 8; ++i) { r[i] = lo[i]; r[i + 8] = hi[i]; }
    return r;
}
// B 32x16 f16 from [N][K] storage: lane n = l&15, K base (l>>4)*16, 16 contiguous halves
__device__ __forceinline__ v16h ldB(const _Float16* p) {
    v8h lo = *(const v8h*)(p);
    v8h hi = *(const v8h*)(p + 8);
    v16h r;
#pragma unroll
    for (int i = 0; i < 8; ++i) { r[i] = lo[i]; r[i + 8] = hi[i]; }
    return r;
}

// ---------------- async global->LDS weight panel (CDNA5 path) ----------------
// Each thread streams its 128B K-slice of column `tid` into its padded LDS row.
// IOFFSET is added to BOTH the LDS and global address, so 8 b128 copies with
// offset:0..112 move the contiguous slice. Tracked by ASYNCcnt.
__device__ __forceinline__ void panel_async_issue(const _Float16* BTg, int kdim,
                                                  _Float16* panel, int tid, int kp) {
    unsigned ldsb = (unsigned)(uintptr_t)(panel + tid * BPS);
    unsigned voff = (unsigned)(((unsigned)tid * (unsigned)kdim + (unsigned)kp) * 2u);
    unsigned long long sbase = (unsigned long long)(uintptr_t)BTg;
    asm volatile(
        "global_load_async_to_lds_b128 %0, %1, %2 offset:0\n"
        "global_load_async_to_lds_b128 %0, %1, %2 offset:16\n"
        "global_load_async_to_lds_b128 %0, %1, %2 offset:32\n"
        "global_load_async_to_lds_b128 %0, %1, %2 offset:48\n"
        "global_load_async_to_lds_b128 %0, %1, %2 offset:64\n"
        "global_load_async_to_lds_b128 %0, %1, %2 offset:80\n"
        "global_load_async_to_lds_b128 %0, %1, %2 offset:96\n"
        "global_load_async_to_lds_b128 %0, %1, %2 offset:112\n"
        :: "v"(ldsb), "v"(voff), "s"(sbase) : "memory");
}

// ---------------- one GEMM stage: O = act(A @ W + b), all f16 in LDS ----------------
// ACT: 0 = fuzzy, 1 = sigmoid (relu(sigmoid(x)) == sigmoid(x))
template <int KDIM, int ACT>
__device__ __forceinline__ void gemm_stage(
    const _Float16* A0, const _Float16* A1,      // wave slabs (stride RS); A1 for k>=256
    const _Float16* __restrict__ BTg,            // global f16 [256][KDIM]
    _Float16* BP,                                // LDS weight panels: 2 x [256][BPS]
    _Float16* O,                                 // wave out slab (stride RS)
    const float* __restrict__ bvec,
    const float* __restrict__ muv,
    const float* __restrict__ sgv,
    int tid) {
    const int lane = tid & 31;
    const int hi   = lane >> 4;
    const int m    = lane & 15;
    const int kb   = hi * 8;
    const int kb2  = hi * 16;

    v8f acc[16];
#pragma unroll
    for (int nt = 0; nt < 16; ++nt)
#pragma unroll
        for (int j = 0; j < 8; ++j) acc[nt][j] = 0.f;

    // prologue: start streaming panel 0
    panel_async_issue(BTg, KDIM, BP, tid, 0);

    for (int kp = 0; kp < KDIM; kp += 64) {
        _Float16* cur = BP + ((kp >> 6) & 1) * PANEL;
        if (kp + 64 < KDIM) {
            _Float16* nxt = BP + (((kp >> 6) + 1) & 1) * PANEL;
            panel_async_issue(BTg, KDIM, nxt, tid, kp + 64);
            asm volatile("s_wait_asynccnt 8" ::: "memory");  // current panel landed
        } else {
            asm volatile("s_wait_asynccnt 0" ::: "memory");
        }
        __syncthreads();   // all waves' slices of `cur` visible

        const _Float16* Arow = ((KDIM == 512 && kp >= 256) ? A1 : A0) + m * RS + (kp & 255);
#pragma unroll
        for (int kk = 0; kk < 64; kk += 32) {
            v16h a = ldA(Arow + kk + kb2);
#pragma unroll
            for (int nt = 0; nt < 16; ++nt) {
                v16h b = ldB(cur + (nt * 16 + m) * BPS + kk + kb2);
                acc[nt] = __builtin_amdgcn_wmma_f32_16x16x32_f16(
                    false, a, false, b, (short)0, acc[nt], false, false);
            }
        }
        __syncthreads();   // done reading `cur` before it is overwritten
    }

    // epilogue: bias + activation, D layout: VGPR j -> row j + 8*hi, col nt*16+m
#pragma unroll
    for (int nt = 0; nt < 16; ++nt) {
        const int col = nt * 16 + m;
        const float bc = bvec[col];
        float mu = 0.f, isg = 0.f;
        if (ACT == 0) { mu = muv[col]; float sg = sgv[col]; isg = 1.f / (sg * sg); }
#pragma unroll
        for (int j = 0; j < 8; ++j) {
            float v = acc[nt][j] + bc;
            float r;
            if (ACT == 0) {
                float z = (v - mu) * isg;
                r = (z >= 0.f) ? __expf(-sqrtf(z)) : v;
            } else {
                r = 1.f / (1.f + __expf(-v));
            }
            O[(j + 8 * hi) * RS + col] = (_Float16)r;
        }
    }
}

// ---------------- fused main kernel: 128 rows/block, 8 waves x 16 rows ----------------
__global__ __launch_bounds__(256, 1) void k_fused(
    const float* __restrict__ x,
    const float* __restrict__ b_fi, const float* __restrict__ mu,
    const float* __restrict__ sigma,
    const float* __restrict__ b1, const float* __restrict__ b2,
    const float* __restrict__ b_fu,
    const float* __restrict__ wsf,
    const _Float16* __restrict__ wfiT, const _Float16* __restrict__ w1T,
    const _Float16* __restrict__ w2T,  const _Float16* __restrict__ wfuT,
    const _Float16* __restrict__ woT,
    float* __restrict__ out) {
    extern __shared__ _Float16 smem[];
    _Float16* XN = smem;                 // 128 x RS
    _Float16* FZ = smem + 128 * RS;      // 128 x RS
    _Float16* HB = smem + 2 * 128 * RS;  // 128 x RS
    _Float16* BP = smem + 3 * 128 * RS;  // 2 x 256 x BPS (double-buffered panel)

    const int tid  = threadIdx.x;
    const int wave = tid >> 5;
    const int lane = tid & 31;
    const int r0   = blockIdx.x * 128;
    const float* scale = wsf + 512;
    const float* shift = wsf + 768;
    const float* bo16  = wsf + 1024;

    // phase 0: load x tile, batchnorm, -> f16 LDS
    {
        const float4* xr = (const float4*)(x + (size_t)r0 * DDIM);
#pragma unroll
        for (int i = 0; i < 32; ++i) {
            int idx = tid + i * 256;            // float4 index in 128x256 tile
            float4 v = xr[idx];
            int r = idx >> 6, c = (idx & 63) * 4;
            float4 s4 = *(const float4*)(scale + c);
            float4 t4 = *(const float4*)(shift + c);
            v4h h;
            h[0] = (_Float16)(v.x * s4.x + t4.x);
            h[1] = (_Float16)(v.y * s4.y + t4.y);
            h[2] = (_Float16)(v.z * s4.z + t4.z);
            h[3] = (_Float16)(v.w * s4.w + t4.w);
            *(v4h*)(XN + r * RS + c) = h;
        }
    }
    __syncthreads();

    const _Float16* myXN = XN + wave * SLAB;
    _Float16* myXNw = XN + wave * SLAB;
    _Float16* myFZ  = FZ + wave * SLAB;
    _Float16* myHB  = HB + wave * SLAB;

    // fz = fuzzy(xn @ W_fi + b_fi)
    gemm_stage<256, 0>(myXN, myXN, wfiT, BP, myFZ, b_fi, mu, sigma, tid);
    // h1 = sigmoid(xn @ W1 + b1)
    gemm_stage<256, 1>(myXN, myXN, w1T, BP, myHB, b1, nullptr, nullptr, tid);
    // h2 = sigmoid(h1 @ W2 + b2)  (overwrites xn slab)
    gemm_stage<256, 1>(myHB, myHB, w2T, BP, myXNw, b2, nullptr, nullptr, tid);
    // fused = sigmoid([fz | h2] @ W_fu + b_fu)  (overwrites h1 slab)
    gemm_stage<512, 1>(myFZ, myXNw, wfuT, BP, myHB, b_fu, nullptr, nullptr, tid);

    // output layer (256 -> 8, padded to 16) + softmax
    {
        const _Float16* F = myHB;
        const int hi = lane >> 4, m = lane & 15;
        const int kb = hi * 8, kb2 = hi * 16;
        v8f acc;
#pragma unroll
        for (int j = 0; j < 8; ++j) acc[j] = 0.f;
#pragma unroll
        for (int kp = 0; kp < 256; kp += 32) {
            v16h a = ldA(F + m * RS + kp + kb2);
            v16h b = ldB(woT + m * 256 + kp + kb2);
            acc = __builtin_amdgcn_wmma_f32_16x16x32_f16(
                false, a, false, b, (short)0, acc, false, false);
        }
        const float bo = bo16[m];
        const int rbase = r0 + wave * 16 + 8 * hi;
#pragma unroll
        for (int j = 0; j < 8; ++j) {
            float v = acc[j] + bo;
            float mx = v;
            mx = fmaxf(mx, __shfl_xor(mx, 1, 32));
            mx = fmaxf(mx, __shfl_xor(mx, 2, 32));
            mx = fmaxf(mx, __shfl_xor(mx, 4, 32));
            float e = __expf(v - mx);
            float s = e;
            s += __shfl_xor(s, 1, 32);
            s += __shfl_xor(s, 2, 32);
            s += __shfl_xor(s, 4, 32);
            if (m < CN) out[(size_t)(rbase + j) * CN + m] = e / s;
        }
    }
}

// ---------------- launcher ----------------
extern "C" void kernel_launch(void* const* d_in, const int* in_sizes, int n_in,
                              void* d_out, int out_size, void* d_ws, size_t ws_size,
                              hipStream_t stream) {
    (void)in_sizes; (void)n_in; (void)out_size; (void)ws_size;
    const float* x     = (const float*)d_in[0];
    const float* gamma = (const float*)d_in[1];
    const float* beta  = (const float*)d_in[2];
    const float* W_fi  = (const float*)d_in[3];
    const float* b_fi  = (const float*)d_in[4];
    const float* mu    = (const float*)d_in[5];
    const float* sigma = (const float*)d_in[6];
    const float* W1    = (const float*)d_in[7];
    const float* b1    = (const float*)d_in[8];
    const float* W2    = (const float*)d_in[9];
    const float* b2    = (const float*)d_in[10];
    const float* W_fu  = (const float*)d_in[11];
    const float* b_fu  = (const float*)d_in[12];
    const float* W_o   = (const float*)d_in[13];
    const float* b_o   = (const float*)d_in[14];
    float* out = (float*)d_out;

    float* wsf = (float*)d_ws;
    _Float16* hbase = (_Float16*)((char*)d_ws + 4352);
    _Float16* wfiT = hbase;
    _Float16* w1T  = hbase + 65536;
    _Float16* w2T  = hbase + 131072;
    _Float16* wfuT = hbase + 196608;   // 256 x 512
    _Float16* woT  = hbase + 327680;   // 16 x 256

    k_zero_stats<<<1, 512, 0, stream>>>(wsf);
    k_colstats<<<256, 256, 0, stream>>>(x, wsf);
    k_finalize<<<1, 256, 0, stream>>>(gamma, beta, b_o, wsf);
    k_transpose<<<256, 256, 0, stream>>>(W_fi, wfiT, 256, 256, 65536);
    k_transpose<<<256, 256, 0, stream>>>(W1,  w1T, 256, 256, 65536);
    k_transpose<<<256, 256, 0, stream>>>(W2,  w2T, 256, 256, 65536);
    k_transpose<<<512, 256, 0, stream>>>(W_fu, wfuT, 512, 256, 131072);
    k_prep_wo<<<16, 256, 0, stream>>>(W_o, woT);

    const size_t shmem = (size_t)(3 * 128 * RS + 2 * PANEL) * sizeof(_Float16); // ~270 KB
    k_fused<<<BATCH / 128, 256, shmem, stream>>>(x, b_fi, mu, sigma, b1, b2, b_fu,
                                                 wsf, wfiT, w1T, w2T, wfuT, woT, out);
}
